// AttentionDecoder_27968827032272
// MI455X (gfx1250) — matrix operand order repaired and checked by reading, not compiled, not run on hardware
//
#include <hip/hip_runtime.h>
#include <hip/hip_bf16.h>
#include <math.h>

// ---------------------------------------------------------------------------
// AttentionDecoder for MI455X (gfx1250, wave32, WMMA).
// Phase 1 (serial over T): GRU0 / GRU1 / attention / W_out via bf16 WMMA.
// Phase 2 (parallel):      [1024 x 1024] x [1024 x 32000] generator GEMM
//                          (LDS-staged A tile, 16x128 per wave) + bias into
//                          d_out, then in-place log_softmax.
// ---------------------------------------------------------------------------

typedef __bf16 bf16_t;
typedef bf16_t v16bf __attribute__((ext_vector_type(16)));
typedef bf16_t v8bf  __attribute__((ext_vector_type(8)));
typedef float  v8f   __attribute__((ext_vector_type(8)));

#define DEV_INLINE __device__ __forceinline__

constexpr int B_ = 32;
constexpr int T_ = 32;
constexpr int S_ = 64;
constexpr int H_ = 1024;
constexpr int E_ = 512;
constexpr int V_ = 32000;
constexpr int BH_ = B_ * H_;

// ---- helpers --------------------------------------------------------------

DEV_INLINE unsigned short f2bf(float f) {
  unsigned u = __float_as_uint(f);
  u += 0x7FFFu + ((u >> 16) & 1u);  // round-to-nearest-even
  return (unsigned short)(u >> 16);
}

DEV_INLINE v8f zero8() {
  v8f z;
#pragma unroll
  for (int i = 0; i < 8; ++i) z[i] = 0.0f;
  return z;
}

// A-fragment: 16x32 bf16, row-major source [rows, ld].
// lanes 0-15: row = lane,    K = k0+0..7   and k0+16..23
// lanes 16-31: row = lane-16, K = k0+8..15 and k0+24..31
DEV_INLINE v16bf load_frag_a(const unsigned short* base, int ld, int m0, int k0, int lane) {
  int row  = m0 + (lane & 15);
  int kofs = k0 + ((lane >> 4) << 3);
  const unsigned short* p = base + (size_t)row * ld + kofs;
  v8bf lo = *reinterpret_cast<const v8bf*>(p);
  v8bf hi = *reinterpret_cast<const v8bf*>(p + 16);
  v16bf a;
#pragma unroll
  for (int i = 0; i < 8; ++i) { a[i] = lo[i]; a[i + 8] = hi[i]; }
  return a;
}

// B-fragment: 32x16 bf16 where B[k][n] = W[n][k], W row-major [N, ld] (x @ W^T).
// lane holds column n = lane&15, K = k0+(lane>>4)*8+0..7 and k0+16+(lane>>4)*8+0..7.
DEV_INLINE v16bf load_frag_b(const unsigned short* base, int ld, int n0, int k0, int lane) {
  int col  = n0 + (lane & 15);
  int kofs = k0 + ((lane >> 4) << 3);
  const unsigned short* p = base + (size_t)col * ld + kofs;
  v8bf lo = *reinterpret_cast<const v8bf*>(p);
  v8bf hi = *reinterpret_cast<const v8bf*>(p + 16);
  v16bf b;
#pragma unroll
  for (int i = 0; i < 8; ++i) { b[i] = lo[i]; b[i + 8] = hi[i]; }
  return b;
}

DEV_INLINE v8f wmma_bf16(v16bf a, v16bf b, v8f c) {
  return __builtin_amdgcn_wmma_f32_16x16x32_bf16(false, a, false, b, (short)0, c, false, false);
}

DEV_INLINE float sigmoidf_(float x) { return 1.0f / (1.0f + expf(-x)); }

// ---- conversion / setup kernels ------------------------------------------

__global__ void f32_to_bf16_k(const float* __restrict__ src, unsigned short* __restrict__ dst, int n) {
  int i = blockIdx.x * blockDim.x + threadIdx.x;
  if (i < n) dst[i] = f2bf(src[i]);
}

// W_a is used un-transposed (q = h1 @ W_a); store it transposed so the
// B-fragment load is contiguous:  dst[n*H + k] = W_a[k*H + n].
__global__ void conv_transpose_k(const float* __restrict__ src, unsigned short* __restrict__ dst) {
  int i = blockIdx.x * blockDim.x + threadIdx.x;
  if (i >= H_ * H_) return;
  int n = i / H_, k = i % H_;
  dst[(size_t)n * H_ + k] = f2bf(src[(size_t)k * H_ + n]);
}

// emb[t][b][e] = word_emb[word_ids[b][t]] + spec_emb[special_ids[b][t]]  (bf16)
__global__ void embed_k(const int* __restrict__ wid, const int* __restrict__ sid,
                        const float* __restrict__ wemb, const float* __restrict__ semb,
                        unsigned short* __restrict__ embb) {
  int i = blockIdx.x * blockDim.x + threadIdx.x;
  if (i >= T_ * B_ * E_) return;
  int e = i % E_;
  int b = (i / E_) % B_;
  int t = i / (E_ * B_);
  int w = wid[b * T_ + t];
  int s = sid[b * T_ + t];
  embb[i] = f2bf(wemb[(size_t)w * E_ + e] + semb[(size_t)s * E_ + e]);
}

__global__ void init_state_k(const float* __restrict__ hidden, const float* __restrict__ prev_out,
                             float* h0, float* h1, float* out,
                             unsigned short* h0b, unsigned short* h1b, unsigned short* outb) {
  int i = blockIdx.x * blockDim.x + threadIdx.x;
  if (i >= BH_) return;
  float a = hidden[i], b = hidden[BH_ + i], c = prev_out[i];
  h0[i] = a; h1[i] = b; out[i] = c;
  h0b[i] = f2bf(a); h1b[i] = f2bf(b); outb[i] = f2bf(c);
}

__global__ void finalize_k(const float* __restrict__ h0, const float* __restrict__ h1,
                           const float* __restrict__ out,
                           float* __restrict__ d_hid, float* __restrict__ d_outvec) {
  int i = blockIdx.x * blockDim.x + threadIdx.x;
  if (i >= BH_) return;
  d_hid[i] = h0[i];
  d_hid[BH_ + i] = h1[i];
  d_outvec[i] = out[i];
}

// ---- GRU cell (fused two GEMMs + gates), one wave per 16x16 N-chunk -------
// grid = (H/16, 2), block = 32 (one wave32; EXEC all ones around WMMA).
// A (x) may be the concat of two row-major sources split at K = splitK.
__global__ __launch_bounds__(32)
void gru_cell_k(const unsigned short* __restrict__ xa, int lda,
                const unsigned short* __restrict__ xb, int ldb, int splitK, int K,
                const unsigned short* __restrict__ w_ih, const unsigned short* __restrict__ w_hh,
                const float* __restrict__ b_ih, const float* __restrict__ b_hh,
                const unsigned short* __restrict__ h_bf, const float* __restrict__ h_prev,
                float* __restrict__ h_out, unsigned short* __restrict__ h_out_bf) {
  const int lane = threadIdx.x;
  const int n0 = blockIdx.x * 16;
  const int m0 = blockIdx.y * 16;

  v8f air = zero8(), aiz = zero8(), ain = zero8();
  v8f ahr = zero8(), ahz = zero8(), ahn = zero8();

  // gi = x @ w_ih^T  (three gate column-tiles at n0, H+n0, 2H+n0)
  for (int k0 = 0; k0 < K; k0 += 32) {
    const unsigned short* ab;
    int ald, kk;
    if (k0 < splitK) { ab = xa; ald = lda; kk = k0; }
    else             { ab = xb; ald = ldb; kk = k0 - splitK; }
    v16bf a  = load_frag_a(ab, ald, m0, kk, lane);
    v16bf br = load_frag_b(w_ih, K, n0,           k0, lane);
    v16bf bz = load_frag_b(w_ih, K, H_ + n0,      k0, lane);
    v16bf bn = load_frag_b(w_ih, K, 2 * H_ + n0,  k0, lane);
    air = wmma_bf16(a, br, air);
    aiz = wmma_bf16(a, bz, aiz);
    ain = wmma_bf16(a, bn, ain);
  }
  // gh = h @ w_hh^T
  for (int k0 = 0; k0 < H_; k0 += 32) {
    v16bf a  = load_frag_a(h_bf, H_, m0, k0, lane);
    v16bf br = load_frag_b(w_hh, H_, n0,          k0, lane);
    v16bf bz = load_frag_b(w_hh, H_, H_ + n0,     k0, lane);
    v16bf bn = load_frag_b(w_hh, H_, 2 * H_ + n0, k0, lane);
    ahr = wmma_bf16(a, br, ahr);
    ahz = wmma_bf16(a, bz, ahz);
    ahn = wmma_bf16(a, bn, ahn);
  }

  // gates + state update; D layout: vgpr v -> M = v + 8*(lane>=16), N = lane&15
#pragma unroll
  for (int v = 0; v < 8; ++v) {
    int row = m0 + v + ((lane >> 4) << 3);
    int col = n0 + (lane & 15);
    float ir = air[v] + b_ih[col];
    float iz = aiz[v] + b_ih[H_ + col];
    float in_ = ain[v] + b_ih[2 * H_ + col];
    float hr = ahr[v] + b_hh[col];
    float hz = ahz[v] + b_hh[H_ + col];
    float hn = ahn[v] + b_hh[2 * H_ + col];
    float r = sigmoidf_(ir + hr);
    float z = sigmoidf_(iz + hz);
    float n = tanhf(in_ + r * hn);
    float hp = h_prev[(size_t)row * H_ + col];
    float h = (1.0f - z) * n + z * hp;
    h_out[(size_t)row * H_ + col] = h;
    h_out_bf[(size_t)row * H_ + col] = f2bf(h);
  }
}

// ---- attention q = h1 @ W_a (W_a pre-transposed) --------------------------
__global__ __launch_bounds__(32)
void attn_q_k(const unsigned short* __restrict__ h1b, const unsigned short* __restrict__ WaT,
              float* __restrict__ q) {
  const int lane = threadIdx.x;
  const int n0 = blockIdx.x * 16;
  const int m0 = blockIdx.y * 16;
  v8f acc = zero8();
  for (int k0 = 0; k0 < H_; k0 += 32) {
    v16bf a = load_frag_a(h1b, H_, m0, k0, lane);
    v16bf b = load_frag_b(WaT, H_, n0, k0, lane);
    acc = wmma_bf16(a, b, acc);
  }
#pragma unroll
  for (int v = 0; v < 8; ++v) {
    int row = m0 + v + ((lane >> 4) << 3);
    int col = n0 + (lane & 15);
    q[(size_t)row * H_ + col] = acc[v];
  }
}

// ---- scores / softmax / context, builds cat = [c | h1] in bf16 ------------
// 4 threads cooperate on each of the S=64 score dots (H=1024 -> 256 each).
__global__ __launch_bounds__(256)
void attn_softmax_ctx_k(const float* __restrict__ q, const float* __restrict__ context,
                        const unsigned char* __restrict__ mask, const float* __restrict__ h1,
                        unsigned short* __restrict__ catb) {
  __shared__ float sc[S_];
  __shared__ float part[256];
  const int b = blockIdx.x, tid = threadIdx.x;
  {
    const int s   = tid >> 2;   // 0..63
    const int sub = tid & 3;    // 0..3
    const float* qb = q + (size_t)b * H_;
    const float* cb = context + ((size_t)b * S_ + s) * H_;
    float acc = 0.0f;
    const int k0 = sub * (H_ / 4);
    for (int k = k0; k < k0 + H_ / 4; ++k) acc += qb[k] * cb[k];
    part[tid] = acc;
  }
  __syncthreads();
  if (tid < S_) {
    float s4 = part[4 * tid] + part[4 * tid + 1] + part[4 * tid + 2] + part[4 * tid + 3];
    sc[tid] = mask[b * S_ + tid] ? s4 : -3.4028235e38f;
  }
  __syncthreads();
  if (tid == 0) {
    float m = -3.4028235e38f;
    for (int s2 = 0; s2 < S_; ++s2) m = fmaxf(m, sc[s2]);
    float sum = 0.0f;
    for (int s2 = 0; s2 < S_; ++s2) { sc[s2] = expf(sc[s2] - m); sum += sc[s2]; }
    float inv = 1.0f / sum;
    for (int s2 = 0; s2 < S_; ++s2) sc[s2] *= inv;
  }
  __syncthreads();
  for (int j = tid; j < H_; j += 256) {
    float acc = 0.0f;
    for (int s2 = 0; s2 < S_; ++s2) acc += sc[s2] * context[((size_t)b * S_ + s2) * H_ + j];
    catb[(size_t)b * 2 * H_ + j]      = f2bf(acc);
    catb[(size_t)b * 2 * H_ + H_ + j] = f2bf(h1[(size_t)b * H_ + j]);
  }
}

// ---- out = tanh(cat @ W_out^T); stores fp32 + bf16 + row into out_all -----
__global__ __launch_bounds__(32)
void attn_out_k(const unsigned short* __restrict__ catb, const unsigned short* __restrict__ Wo,
                float* __restrict__ out_f, unsigned short* __restrict__ out_b,
                unsigned short* __restrict__ out_all_row) {
  const int lane = threadIdx.x;
  const int n0 = blockIdx.x * 16;
  const int m0 = blockIdx.y * 16;
  v8f acc = zero8();
  for (int k0 = 0; k0 < 2 * H_; k0 += 32) {
    v16bf a = load_frag_a(catb, 2 * H_, m0, k0, lane);
    v16bf b = load_frag_b(Wo,   2 * H_, n0, k0, lane);
    acc = wmma_bf16(a, b, acc);
  }
#pragma unroll
  for (int v = 0; v < 8; ++v) {
    int row = m0 + v + ((lane >> 4) << 3);
    int col = n0 + (lane & 15);
    float o = tanhf(acc[v]);
    out_f[(size_t)row * H_ + col] = o;
    unsigned short ob = f2bf(o);
    out_b[(size_t)row * H_ + col] = ob;
    out_all_row[(size_t)row * H_ + col] = ob;
  }
}

// ---- generator: logits = out_all @ W_gen^T + b_gen ------------------------
// grid = (V/256, 1024/16), block = 64 (2 waves, each wave a 16x128 tile).
// The block's 16-row A tile (full K) is one contiguous 32KB slab -> staged
// once into LDS and shared by both waves; B streams from global (L2-resident).
__global__ __launch_bounds__(64)
void gen_logits_k(const unsigned short* __restrict__ Aall, const unsigned short* __restrict__ Wg,
                  const float* __restrict__ bg, float* __restrict__ logits) {
  __shared__ unsigned short As[16 * H_];  // 32 KB
  const int tid  = threadIdx.x;
  const int lane = tid & 31;
  const int wave = tid >> 5;
  const int m0 = blockIdx.y * 16;
  const int n0 = blockIdx.x * 256 + wave * 128;

  // cooperative copy: rows m0..m0+15 of A over full K are contiguous
  {
    const uint4* src = reinterpret_cast<const uint4*>(Aall + (size_t)m0 * H_);
    uint4* dst = reinterpret_cast<uint4*>(As);
    for (int i = tid; i < 16 * H_ / 8; i += 64) dst[i] = src[i];
  }
  __syncthreads();

  v8f acc[8];
#pragma unroll
  for (int i = 0; i < 8; ++i) acc[i] = zero8();

  for (int k0 = 0; k0 < H_; k0 += 32) {
    if (k0 + 32 < H_)
      __builtin_prefetch(Wg + (size_t)(n0 + (lane & 15)) * H_ + k0 + 32, 0, 1);
    v16bf a = load_frag_a(As, H_, 0, k0, lane);  // ds_load from LDS
#pragma unroll
    for (int i = 0; i < 8; ++i) {
      v16bf b = load_frag_b(Wg, H_, n0 + 16 * i, k0, lane);
      acc[i] = wmma_bf16(a, b, acc[i]);
    }
  }

#pragma unroll
  for (int i = 0; i < 8; ++i) {
#pragma unroll
    for (int v = 0; v < 8; ++v) {
      int row = m0 + v + ((lane >> 4) << 3);
      int col = n0 + 16 * i + (lane & 15);
      logits[(size_t)row * V_ + col] = acc[i][v] + bg[col];
    }
  }
}

// ---- in-place row log_softmax over V --------------------------------------
__global__ __launch_bounds__(256)
void log_softmax_k(float* __restrict__ logits) {
  __shared__ float red[256];
  float* row = logits + (size_t)blockIdx.x * V_;
  const int tid = threadIdx.x;
  float m = -3.4028235e38f;
  for (int j = tid; j < V_; j += 256) m = fmaxf(m, row[j]);
  red[tid] = m; __syncthreads();
  for (int s = 128; s > 0; s >>= 1) { if (tid < s) red[tid] = fmaxf(red[tid], red[tid + s]); __syncthreads(); }
  m = red[0];
  __syncthreads();
  float sum = 0.0f;
  for (int j = tid; j < V_; j += 256) sum += expf(row[j] - m);
  red[tid] = sum; __syncthreads();
  for (int s = 128; s > 0; s >>= 1) { if (tid < s) red[tid] += red[tid + s]; __syncthreads(); }
  float lse = m + logf(red[0]);
  for (int j = tid; j < V_; j += 256) row[j] -= lse;
}

// ---------------------------------------------------------------------------

extern "C" void kernel_launch(void* const* d_in, const int* in_sizes, int n_in,
                              void* d_out, int out_size, void* d_ws, size_t ws_size,
                              hipStream_t stream) {
  (void)in_sizes; (void)n_in; (void)out_size; (void)ws_size;

  const int*           word_ids    = (const int*)d_in[0];
  const int*           special_ids = (const int*)d_in[1];
  const unsigned char* ctx_mask    = (const unsigned char*)d_in[2];  // bool
  const float* word_emb = (const float*)d_in[3];
  const float* spec_emb = (const float*)d_in[4];
  const float* w_ih0 = (const float*)d_in[5];
  const float* w_hh0 = (const float*)d_in[6];
  const float* b_ih0 = (const float*)d_in[7];
  const float* b_hh0 = (const float*)d_in[8];
  const float* w_ih1 = (const float*)d_in[9];
  const float* w_hh1 = (const float*)d_in[10];
  const float* b_ih1 = (const float*)d_in[11];
  const float* b_hh1 = (const float*)d_in[12];
  const float* W_a   = (const float*)d_in[13];
  const float* W_out = (const float*)d_in[14];
  const float* W_gen = (const float*)d_in[15];
  const float* b_gen = (const float*)d_in[16];
  const float* hidden      = (const float*)d_in[17];
  const float* prev_output = (const float*)d_in[18];
  const float* context     = (const float*)d_in[19];

  float* out_scores = (float*)d_out;                       // [T,B,V]
  float* out_hid    = out_scores + (size_t)T_ * B_ * V_;   // [2,B,H]
  float* out_vec    = out_hid + 2 * (size_t)BH_;           // [B,H]

  // ---- workspace carve (256B aligned) ----
  char* wp = (char*)d_ws;
  auto alloc = [&](size_t bytes) -> void* {
    void* r = (void*)wp;
    wp += (bytes + 255) & ~(size_t)255;
    return r;
  };
  typedef unsigned short u16;
  u16* w_ih0_bf = (u16*)alloc(sizeof(u16) * (size_t)3 * H_ * (E_ + H_));
  u16* w_hh0_bf = (u16*)alloc(sizeof(u16) * (size_t)3 * H_ * H_);
  u16* w_ih1_bf = (u16*)alloc(sizeof(u16) * (size_t)3 * H_ * H_);
  u16* w_hh1_bf = (u16*)alloc(sizeof(u16) * (size_t)3 * H_ * H_);
  u16* WaT_bf   = (u16*)alloc(sizeof(u16) * (size_t)H_ * H_);
  u16* Wout_bf  = (u16*)alloc(sizeof(u16) * (size_t)H_ * 2 * H_);
  u16* Wgen_bf  = (u16*)alloc(sizeof(u16) * (size_t)V_ * H_);
  u16* emb_bf   = (u16*)alloc(sizeof(u16) * (size_t)T_ * B_ * E_);
  u16* out_all  = (u16*)alloc(sizeof(u16) * (size_t)T_ * BH_);
  float* h0_f[2], *h1_f[2], *outf[2];
  u16*   h0_b[2], *h1_b[2], *outb[2];
  for (int i = 0; i < 2; ++i) {
    h0_f[i] = (float*)alloc(sizeof(float) * BH_);
    h1_f[i] = (float*)alloc(sizeof(float) * BH_);
    outf[i] = (float*)alloc(sizeof(float) * BH_);
    h0_b[i] = (u16*)alloc(sizeof(u16) * BH_);
    h1_b[i] = (u16*)alloc(sizeof(u16) * BH_);
    outb[i] = (u16*)alloc(sizeof(u16) * BH_);
  }
  float* q_f    = (float*)alloc(sizeof(float) * BH_);
  u16*   cat_bf = (u16*)alloc(sizeof(u16) * (size_t)B_ * 2 * H_);

  // ---- weight conversions (fp32 -> bf16, once per launch) ----
  auto conv = [&](const float* s, u16* d, size_t n) {
    f32_to_bf16_k<<<(int)((n + 255) / 256), 256, 0, stream>>>(s, d, (int)n);
  };
  conv(w_ih0, w_ih0_bf, (size_t)3 * H_ * (E_ + H_));
  conv(w_hh0, w_hh0_bf, (size_t)3 * H_ * H_);
  conv(w_ih1, w_ih1_bf, (size_t)3 * H_ * H_);
  conv(w_hh1, w_hh1_bf, (size_t)3 * H_ * H_);
  conv_transpose_k<<<(H_ * H_ + 255) / 256, 256, 0, stream>>>(W_a, WaT_bf);
  conv(W_out, Wout_bf, (size_t)H_ * 2 * H_);
  conv(W_gen, Wgen_bf, (size_t)V_ * H_);

  // ---- embeddings + state init ----
  embed_k<<<(T_ * B_ * E_ + 255) / 256, 256, 0, stream>>>(word_ids, special_ids,
                                                          word_emb, spec_emb, emb_bf);
  init_state_k<<<(BH_ + 255) / 256, 256, 0, stream>>>(hidden, prev_output,
                                                      h0_f[0], h1_f[0], outf[0],
                                                      h0_b[0], h1_b[0], outb[0]);

  // ---- phase 1: serial scan (ping-pong state buffers) ----
  const dim3 gGemm(H_ / 16, B_ / 16);  // (64, 2)
  int cur = 0;
  for (int t = 0; t < T_; ++t) {
    const int nxt = cur ^ 1;
    const u16* emb_t = emb_bf + (size_t)t * B_ * E_;

    // GRU cell 0: x = [emb_t | out_prev], K = E + H
    gru_cell_k<<<gGemm, 32, 0, stream>>>(emb_t, E_, outb[cur], H_, E_, E_ + H_,
                                         w_ih0_bf, w_hh0_bf, b_ih0, b_hh0,
                                         h0_b[cur], h0_f[cur], h0_f[nxt], h0_b[nxt]);
    // GRU cell 1: x = h0_new, K = H
    gru_cell_k<<<gGemm, 32, 0, stream>>>(h0_b[nxt], H_, nullptr, 0, H_, H_,
                                         w_ih1_bf, w_hh1_bf, b_ih1, b_hh1,
                                         h1_b[cur], h1_f[cur], h1_f[nxt], h1_b[nxt]);
    // attention
    attn_q_k<<<gGemm, 32, 0, stream>>>(h1_b[nxt], WaT_bf, q_f);
    attn_softmax_ctx_k<<<B_, 256, 0, stream>>>(q_f, context, ctx_mask, h1_f[nxt], cat_bf);
    attn_out_k<<<gGemm, 32, 0, stream>>>(cat_bf, Wout_bf, outf[nxt], outb[nxt],
                                         out_all + (size_t)t * BH_);
    cur = nxt;
  }

  // ---- phase 2: generator GEMM + log_softmax ----
  gen_logits_k<<<dim3(V_ / 256, (T_ * B_) / 16), 64, 0, stream>>>(out_all, Wgen_bf,
                                                                  b_gen, out_scores);
  log_softmax_k<<<T_ * B_, 256, 0, stream>>>(out_scores);

  // ---- final states ----
  finalize_k<<<(BH_ + 255) / 256, 256, 0, stream>>>(h0_f[cur], h1_f[cur], outf[cur],
                                                    out_hid, out_vec);
}
